// SchNet_18983755448808
// MI455X (gfx1250) — matrix operand order repaired and checked by reading, not compiled, not run on hardware
//
#include <hip/hip_runtime.h>
#include <hip/hip_bf16.h>

// ---------------- problem constants ----------------
enum { N_ATOM = 16384, E_EDGE = 262144, DIM = 128, FILT = 128, GAUSS = 50,
       GPAD = 64, LAY = 4, NMOL = 32 };

typedef __attribute__((ext_vector_type(16))) _Float16 v16h;
typedef __attribute__((ext_vector_type(8)))  float    v8f;
typedef __attribute__((ext_vector_type(4))) unsigned int u32x4;
typedef __attribute__((ext_vector_type(8))) int i32x8;
typedef __attribute__((ext_vector_type(4))) int i32x4;

__device__ __forceinline__ float sspf(float x) {
    // softplus(x) - log(2), numerically stable
    return fmaxf(x, 0.0f) + __logf(1.0f + __expf(-fabsf(x))) - 0.69314718056f;
}

__device__ __forceinline__ v8f wmma32(v16h a, v16h b, v8f c) {
    return __builtin_amdgcn_wmma_f32_16x16x32_f16(false, a, false, b,
                                                  (short)0, c, false, false);
}

// ---- Tensor Data Mover: DMA a 32 KB contiguous block (global -> LDS) --------
// D# per CDNA5 ISA §8: group0 = {count=1 | lds_addr | global_addr | type=2},
// group1 = 1-D tensor: data_size=8B, tensor_dim0 = tile_dim0 = 4096 units.
__device__ __forceinline__ void tdm_load_32kb(const _Float16* gsrc,
                                              _Float16* lds_dst) {
    unsigned long long ga = (unsigned long long)(size_t)gsrc;
    unsigned int la = (unsigned int)(size_t)(void*)lds_dst;
    u32x4 g0;
    g0[0] = 1u;                                      // count=1 (valid user D#)
    g0[1] = la;                                      // lds_addr [63:32]
    g0[2] = (unsigned int)ga;                        // global_addr [95:64]
    g0[3] = (unsigned int)(ga >> 32) | (2u << 30);   // ga[56:32] | type=2
    i32x8 g1;
    g1[0] = 0x00030000;          // wg_mask=0, data_size=3 (8 B units)
    g1[1] = (int)(4096u << 16);  // tensor_dim0[15:0] = 4096  (bits 63:48)
    g1[2] = (int)(1u << 16);     // tensor_dim0 hi = 0 | tensor_dim1 = 1
    g1[3] = (int)(4096u << 16);  // tensor_dim1 hi = 0 | tile_dim0 = 4096
    g1[4] = 0;                   // tile_dim1 = 0 (1-D), tile_dim2 = 0
    g1[5] = 4096;                // tensor_dim0_stride[31:0]
    g1[6] = 0;
    g1[7] = 0;
    i32x4 gz = {0, 0, 0, 0};
    i32x8 gz8 = {0, 0, 0, 0, 0, 0, 0, 0};
    __builtin_amdgcn_tensor_load_to_lds(g0, g1, gz, gz, gz8, 0);
}

// A fragment (16x32 f16) from row-major [rows, ldk] at (row0, k0).
// ISA 7.12.2: lane<16 -> K 0..7 & 16..23 ; lane>=16 -> K 8..15 & 24..31.
// Per-lane data = two contiguous 8-half (16 B) runs -> two b128 loads.
__device__ __forceinline__ v16h load_a_f16(const _Float16* M, int ldk,
                                           int row0, int k0, int lane) {
    int row = row0 + (lane & 15);
    int kb  = k0 + ((lane >> 4) << 3);
    const _Float16* p = M + (size_t)row * ldk + kb;
    v16h a;
    ((uint4*)&a)[0] = *(const uint4*)(p);
    ((uint4*)&a)[1] = *(const uint4*)(p + 16);
    return a;
}

// Same but f32 source, converted to f16 (vector f32 loads + packed cvt).
__device__ __forceinline__ v16h load_a_f32cvt(const float* M, int ldk,
                                              int row0, int k0, int lane) {
    int row = row0 + (lane & 15);
    int kb  = k0 + ((lane >> 4) << 3);
    const float* p = M + (size_t)row * ldk + kb;
    float4 f0 = ((const float4*)p)[0];
    float4 f1 = ((const float4*)p)[1];
    float4 f2 = ((const float4*)(p + 16))[0];
    float4 f3 = ((const float4*)(p + 16))[1];
    v16h a;
    a[0]  = (_Float16)f0.x; a[1]  = (_Float16)f0.y;
    a[2]  = (_Float16)f0.z; a[3]  = (_Float16)f0.w;
    a[4]  = (_Float16)f1.x; a[5]  = (_Float16)f1.y;
    a[6]  = (_Float16)f1.z; a[7]  = (_Float16)f1.w;
    a[8]  = (_Float16)f2.x; a[9]  = (_Float16)f2.y;
    a[10] = (_Float16)f2.z; a[11] = (_Float16)f2.w;
    a[12] = (_Float16)f3.x; a[13] = (_Float16)f3.y;
    a[14] = (_Float16)f3.z; a[15] = (_Float16)f3.w;
    return a;
}

// B fragment from PRE-SWIZZLED fragment-major storage:
//   Bf[((kt*8 + nt)*32 + lane)*16 + i]   (N fixed at 128 -> 8 n-tiles)
// Per-lane 32 contiguous bytes -> two b128 loads, zero packing.
__device__ __forceinline__ v16h load_b_frag(const _Float16* Bf, int kt, int nt,
                                            int lane) {
    const _Float16* p = Bf + (((size_t)(kt * 8 + nt) * 32 + lane) << 4);
    v16h b;
    ((uint4*)&b)[0] = ((const uint4*)p)[0];
    ((uint4*)&b)[1] = ((const uint4*)p)[1];
    return b;
}

// ---------------- simple elementwise kernels ----------------
__global__ void k_zero(float* __restrict__ p, int n) {
    int i = blockIdx.x * blockDim.x + threadIdx.x;
    if (i < n) p[i] = 0.0f;
}

__global__ void k_embed(const float* __restrict__ z, const float* __restrict__ W,
                        const float* __restrict__ b, float* __restrict__ h,
                        _Float16* __restrict__ h16) {
    int i = blockIdx.x * blockDim.x + threadIdx.x;   // N*128 threads
    int n = i >> 7, d = i & 127;
    float v = z[n] * W[d] + b[d];
    h[i] = v;
    h16[i] = (_Float16)v;
}

__global__ void k_edge(const float* __restrict__ pos, const int* __restrict__ esrc,
                       const int* __restrict__ edst, _Float16* __restrict__ rbf16,
                       float* __restrict__ Cenv) {
    int e = blockIdx.x * blockDim.x + threadIdx.x;
    if (e >= E_EDGE) return;
    int s = esrc[e], d = edst[e];
    float dx = pos[s * 3 + 0] - pos[d * 3 + 0];
    float dy = pos[s * 3 + 1] - pos[d * 3 + 1];
    float dz = pos[s * 3 + 2] - pos[d * 3 + 2];
    float dist = sqrtf(dx * dx + dy * dy + dz * dz + 1e-12f);
    Cenv[e] = 0.5f * (__cosf(dist * (3.14159265358979f / 5.0f)) + 1.0f);
    _Float16* r = rbf16 + (size_t)e * GPAD;
#pragma unroll
    for (int g = 0; g < GPAD; ++g) {
        float mu = (5.0f / 49.0f) * (float)g;
        float t = dist - mu;
        float v = (g < GAUSS) ? __expf(-4.0f * t * t) : 0.0f;
        r[g] = (_Float16)v;
    }
}

// f32 [L, Kvalid, 128] row-major  ->  f16 fragment-major [L, Ktot/32, 8, 32, 16]
// (rows Kvalid..Ktot-1 zero-padded)
__global__ void k_cvt_swz(const float* __restrict__ in, _Float16* __restrict__ out,
                          int Ktot, int Kvalid, int strideIn, int strideOut) {
    int i = blockIdx.x * blockDim.x + threadIdx.x;   // LAY * Ktot * 128
    int per = Ktot * 128;
    int l = i / per, rem = i % per;
    int k = rem >> 7, n = rem & 127;
    float v = (k < Kvalid) ? in[(size_t)l * strideIn + k * 128 + n] : 0.0f;
    int kt = k >> 5, kin = k & 31, nt = n >> 4, nin = n & 15;
    int lane = nin | (((kin >> 3) & 1) << 4);
    int ii = (kin & 7) | ((kin >> 4) << 3);
    out[(size_t)l * strideOut + (((size_t)(kt * 8 + nt) * 32 + lane) << 4) + ii] =
        (_Float16)v;
}

// ---------------- WMMA GEMM: hW16 = f16(h16 @ B[128,128]) ----------------
__global__ __launch_bounds__(128) void k_gemm_f16out(
    const _Float16* __restrict__ A, const _Float16* __restrict__ Bf,
    _Float16* __restrict__ Out) {
    __shared__ _Float16 sB[128 * 128];   // swizzled fragment-major copy
    int tid = threadIdx.x;
    if (tid < 32) {                      // wave 0 drives the TDM
        tdm_load_32kb(Bf, sB);
        __builtin_amdgcn_s_wait_tensorcnt(0);
    }
    __syncthreads();
    int wave = tid >> 5, lane = tid & 31;
    int r0 = blockIdx.x * 64 + wave * 16;
    v16h af[4];
#pragma unroll
    for (int ks = 0; ks < 4; ++ks) af[ks] = load_a_f16(A, 128, r0, ks * 32, lane);
    for (int nt = 0; nt < 8; ++nt) {
        v8f acc = {};
#pragma unroll
        for (int ks = 0; ks < 4; ++ks)
            acc = wmma32(af[ks], load_b_frag(sB, ks, nt, lane), acc);
        int col = lane & 15, rb = (lane >> 4) * 8;
#pragma unroll
        for (int v = 0; v < 8; ++v)
            Out[(size_t)(r0 + rb + v) * 128 + nt * 16 + col] = (_Float16)acc[v];
    }
}

// ---------------- fused CFConv: filter double-GEMM + gather*mul + scatter-add ----
__global__ __launch_bounds__(128) void k_cfconv(
    const _Float16* __restrict__ rbf16, const float* __restrict__ Cenv,
    const _Float16* __restrict__ W1f,   // [2kt,8nt,32,16] f16 swizzled (this layer)
    const float* __restrict__ b1,       // [128]
    const _Float16* __restrict__ W2f,   // [4kt,8nt,32,16] f16 swizzled
    const float* __restrict__ b2,       // [128]
    const _Float16* __restrict__ hW16,  // [N,128] f16
    const int* __restrict__ esrc, const int* __restrict__ edst,
    float* __restrict__ agg) {
    __shared__ _Float16 sW2[128 * 128];     // 32 KB, swizzled (via TDM)
    __shared__ _Float16 stg[4][16 * 128];   // 16 KB per-wave staging (row-major t)
    int tid = threadIdx.x;
    if (tid < 32) {
        tdm_load_32kb(W2f, sW2);
        __builtin_amdgcn_s_wait_tensorcnt(0);
    }
    __syncthreads();

    int wave = tid >> 5, lane = tid & 31;
    int e0 = blockIdx.x * 64 + wave * 16;
    _Float16* my = stg[wave];
    int col = lane & 15, rb = (lane >> 4) * 8;

    // stage 1: t[16,128] = ssp(rbf_tile[16,64] @ W1 + b1)
    v16h a0 = load_a_f16(rbf16, GPAD, e0, 0, lane);
    v16h a1 = load_a_f16(rbf16, GPAD, e0, 32, lane);
    for (int nt = 0; nt < 8; ++nt) {
        v8f acc = {};
        acc = wmma32(a0, load_b_frag(W1f, 0, nt, lane), acc);
        acc = wmma32(a1, load_b_frag(W1f, 1, nt, lane), acc);
        float bb = b1[nt * 16 + col];
#pragma unroll
        for (int v = 0; v < 8; ++v)
            my[(rb + v) * 128 + nt * 16 + col] = (_Float16)sspf(acc[v] + bb);
    }
    asm volatile("s_wait_dscnt 0" ::: "memory");
    __builtin_amdgcn_wave_barrier();

    // stage 2: Wf[16,128] = (t @ W2 + b2) * C ; gather hW16[src]*Wf -> agg[dst]
    for (int nt = 0; nt < 8; ++nt) {
        int n0 = nt * 16;
        v8f acc = {};
#pragma unroll
        for (int ks = 0; ks < 4; ++ks) {
            v16h a = load_a_f16(my, 128, 0, ks * 32, lane);
            acc = wmma32(a, load_b_frag(sW2, ks, nt, lane), acc);
        }
        float bb = b2[n0 + col];
#pragma unroll
        for (int v = 0; v < 8; ++v) {
            int e = e0 + rb + v;
            float wf = (acc[v] + bb) * Cenv[e];
            int s = esrc[e], d = edst[e];
            float xj = (float)hW16[(size_t)s * 128 + n0 + col];
            atomicAdd(&agg[(size_t)d * 128 + n0 + col], xj * wf);
        }
    }
}

// ---------------- fused node update: h += (ssp(agg@cf2+b) @ intW + b) ----------
__global__ __launch_bounds__(128) void k_update(
    const float* __restrict__ agg,       // [N,128] f32
    const _Float16* __restrict__ cf2f,   // swizzled f16 (stage-1 B, via L2)
    const float* __restrict__ cf2b,
    const _Float16* __restrict__ intWf,  // swizzled f16 (stage-2 B, in LDS)
    const float* __restrict__ intb,
    float* __restrict__ h, _Float16* __restrict__ h16) {
    __shared__ _Float16 sB[128 * 128];
    __shared__ _Float16 stg[4][16 * 128];
    int tid = threadIdx.x;
    if (tid < 32) {
        tdm_load_32kb(intWf, sB);
        __builtin_amdgcn_s_wait_tensorcnt(0);
    }
    __syncthreads();

    int wave = tid >> 5, lane = tid & 31;
    int r0 = blockIdx.x * 64 + wave * 16;
    _Float16* my = stg[wave];
    int col = lane & 15, rb = (lane >> 4) * 8;

    // stage 1: x = ssp(agg @ cf2 + cf2b)
    v16h af[4];
#pragma unroll
    for (int ks = 0; ks < 4; ++ks) af[ks] = load_a_f32cvt(agg, 128, r0, ks * 32, lane);
    for (int nt = 0; nt < 8; ++nt) {
        v8f acc = {};
#pragma unroll
        for (int ks = 0; ks < 4; ++ks)
            acc = wmma32(af[ks], load_b_frag(cf2f, ks, nt, lane), acc);
        float bb = cf2b[nt * 16 + col];
#pragma unroll
        for (int v = 0; v < 8; ++v)
            my[(rb + v) * 128 + nt * 16 + col] = (_Float16)sspf(acc[v] + bb);
    }
    asm volatile("s_wait_dscnt 0" ::: "memory");
    __builtin_amdgcn_wave_barrier();

    // stage 2: h += x @ intW + intb
    for (int nt = 0; nt < 8; ++nt) {
        int n0 = nt * 16;
        v8f acc = {};
#pragma unroll
        for (int ks = 0; ks < 4; ++ks) {
            v16h a = load_a_f16(my, 128, 0, ks * 32, lane);
            acc = wmma32(a, load_b_frag(sB, ks, nt, lane), acc);
        }
        float bb = intb[n0 + col];
#pragma unroll
        for (int v = 0; v < 8; ++v) {
            size_t idx = (size_t)(r0 + rb + v) * 128 + n0 + col;
            float hv = h[idx] + acc[v] + bb;
            h[idx] = hv;
            h16[idx] = (_Float16)hv;
        }
    }
}

// ---------------- readout head + per-molecule reduction ----------------
__global__ __launch_bounds__(64) void k_head(
    const float* __restrict__ h, const float* __restrict__ W1,
    const float* __restrict__ b1v, const float* __restrict__ W2v,
    const float* __restrict__ b2v, const int* __restrict__ batch,
    float* __restrict__ out) {
    __shared__ float red[64];
    int n = blockIdx.x, t = threadIdx.x;
    const float* hr = h + (size_t)n * 128;
    float s = b1v[t];
    for (int d = 0; d < 128; ++d) s += hr[d] * W1[d * 64 + t];
    red[t] = sspf(s) * W2v[t];
    __syncthreads();
    if (t == 0) {
        float tot = b2v[0];
        for (int i = 0; i < 64; ++i) tot += red[i];
        atomicAdd(&out[batch[n]], tot);
    }
}

// ---------------- host launcher ----------------
extern "C" void kernel_launch(void* const* d_in, const int* in_sizes, int n_in,
                              void* d_out, int out_size, void* d_ws, size_t ws_size,
                              hipStream_t stream) {
    const float* z    = (const float*)d_in[0];
    const float* pos  = (const float*)d_in[1];
    const int*   batch= (const int*)d_in[2];
    const int*   esrc = (const int*)d_in[3];
    const int*   edst = (const int*)d_in[4];
    const float* embW = (const float*)d_in[5];
    const float* embB = (const float*)d_in[6];
    const float* mlpW1= (const float*)d_in[7];
    const float* mlpB1= (const float*)d_in[8];
    const float* mlpW2= (const float*)d_in[9];
    const float* mlpB2= (const float*)d_in[10];
    const float* cf1W = (const float*)d_in[11];
    const float* cf2W = (const float*)d_in[12];
    const float* cf2B = (const float*)d_in[13];
    const float* intW = (const float*)d_in[14];
    const float* intB = (const float*)d_in[15];
    const float* l1W  = (const float*)d_in[16];
    const float* l1B  = (const float*)d_in[17];
    const float* l2W  = (const float*)d_in[18];
    const float* l2B  = (const float*)d_in[19];
    float* out = (float*)d_out;

    char* ws = (char*)d_ws;
    size_t o = 0;
    auto alloc = [&](size_t bytes) { void* p = ws + o; o = (o + bytes + 255) & ~(size_t)255; return p; };
    float*     h      = (float*)    alloc((size_t)N_ATOM * DIM * 4);
    _Float16*  h16    = (_Float16*) alloc((size_t)N_ATOM * DIM * 2);
    _Float16*  hW16   = (_Float16*) alloc((size_t)N_ATOM * FILT * 2);
    float*     agg    = (float*)    alloc((size_t)N_ATOM * FILT * 4);
    _Float16*  rbf16  = (_Float16*) alloc((size_t)E_EDGE * GPAD * 2);
    float*     Cenv   = (float*)    alloc((size_t)E_EDGE * 4);
    _Float16*  w1f    = (_Float16*) alloc((size_t)LAY * GPAD * FILT * 2);
    _Float16*  w2f    = (_Float16*) alloc((size_t)LAY * FILT * FILT * 2);
    _Float16*  cf1f   = (_Float16*) alloc((size_t)LAY * DIM * FILT * 2);
    _Float16*  cf2f   = (_Float16*) alloc((size_t)LAY * FILT * DIM * 2);
    _Float16*  intf   = (_Float16*) alloc((size_t)LAY * DIM * DIM * 2);

    k_embed<<<(N_ATOM * DIM) / 256, 256, 0, stream>>>(z, embW, embB, h, h16);
    k_edge<<<E_EDGE / 256, 256, 0, stream>>>(pos, esrc, edst, rbf16, Cenv);
    // weight converts: f32 row-major -> f16 fragment-major (swizzled)
    k_cvt_swz<<<(LAY * GPAD * FILT) / 256, 256, 0, stream>>>(
        mlpW1, w1f, GPAD, GAUSS, GAUSS * FILT, GPAD * FILT);
    k_cvt_swz<<<(LAY * FILT * FILT) / 256, 256, 0, stream>>>(
        mlpW2, w2f, FILT, FILT, FILT * FILT, FILT * FILT);
    k_cvt_swz<<<(LAY * DIM * FILT) / 256, 256, 0, stream>>>(
        cf1W, cf1f, DIM, DIM, DIM * FILT, DIM * FILT);
    k_cvt_swz<<<(LAY * FILT * DIM) / 256, 256, 0, stream>>>(
        cf2W, cf2f, FILT, FILT, FILT * DIM, FILT * DIM);
    k_cvt_swz<<<(LAY * DIM * DIM) / 256, 256, 0, stream>>>(
        intW, intf, DIM, DIM, DIM * DIM, DIM * DIM);

    for (int l = 0; l < LAY; ++l) {
        k_gemm_f16out<<<N_ATOM / 64, 128, 0, stream>>>(
            h16, cf1f + (size_t)l * DIM * FILT, hW16);
        k_zero<<<(N_ATOM * FILT) / 256, 256, 0, stream>>>(agg, N_ATOM * FILT);
        k_cfconv<<<E_EDGE / 64, 128, 0, stream>>>(
            rbf16, Cenv, w1f + (size_t)l * GPAD * FILT, mlpB1 + l * FILT,
            w2f + (size_t)l * FILT * FILT, mlpB2 + l * FILT, hW16, esrc, edst, agg);
        k_update<<<N_ATOM / 64, 128, 0, stream>>>(
            agg, cf2f + (size_t)l * FILT * DIM, cf2B + l * DIM,
            intf + (size_t)l * DIM * DIM, intB + l * DIM, h, h16);
    }

    k_zero<<<1, 32, 0, stream>>>(out, NMOL);
    k_head<<<N_ATOM, 64, 0, stream>>>(h, l1W, l1B, l2W, l2B, batch, out);
}
